// GNN_6949257085137
// MI455X (gfx1250) — matrix operand order repaired and checked by reading, not compiled, not run on hardware
//
#include <hip/hip_runtime.h>

typedef __bf16 bf16_t;
typedef __attribute__((ext_vector_type(16))) __bf16 v16bf;
typedef __attribute__((ext_vector_type(8)))  float  v8f;

#define FAN_IN   512
#define FAN_MID  256
#define FAN_OUT  40
#define FAN_OUT_P 48   // padded to 3 column tiles of 16
#define LN_EPS   1e-5f

// ---------------------------------------------------------------------------
// Prep: transpose W1 -> bf16 [256][512], W2 -> bf16 [48][256] (zero padded)
// ---------------------------------------------------------------------------
__global__ void prep_w_kernel(const float* __restrict__ W1,
                              const float* __restrict__ W2,
                              bf16_t* __restrict__ w1t,
                              bf16_t* __restrict__ w2t) {
  int i = blockIdx.x * blockDim.x + threadIdx.x;
  if (i < FAN_MID * FAN_IN) {           // 256*512
    int n = i / FAN_IN, k = i % FAN_IN;
    w1t[i] = (bf16_t)W1[k * FAN_MID + n];
  }
  if (i < FAN_OUT_P * FAN_MID) {        // 48*256
    int n = i / FAN_MID, k = i % FAN_MID;
    w2t[i] = (n < FAN_OUT) ? (bf16_t)W2[k * FAN_OUT + n] : (bf16_t)0.0f;
  }
}

// ---------------------------------------------------------------------------
// Zero accumulators, deg = 1 (self loop)
// ---------------------------------------------------------------------------
__global__ void init_kernel(float* __restrict__ agg1, float* __restrict__ agg2,
                            float* __restrict__ deg, int N) {
  size_t i = (size_t)blockIdx.x * blockDim.x + threadIdx.x;
  if (i < (size_t)N * FAN_MID) agg1[i] = 0.0f;
  if (i < (size_t)N * FAN_OUT) agg2[i] = 0.0f;
  if (i < (size_t)N)           deg[i]  = 1.0f;
}

__global__ void deg_kernel(const long long* __restrict__ ei, float* __restrict__ deg,
                           int E) {
  int e = blockIdx.x * blockDim.x + threadIdx.x;
  if (e < E) {
    int d = (int)ei[(size_t)E + e];   // row 1 = dst
    atomicAdd(&deg[d], 1.0f);
  }
}

__global__ void dinv_kernel(const float* __restrict__ deg, float* __restrict__ dinv,
                            int N) {
  int i = blockIdx.x * blockDim.x + threadIdx.x;
  if (i < N) dinv[i] = rsqrtf(deg[i]);
}

__global__ void norm_kernel(const long long* __restrict__ ei,
                            const float* __restrict__ dinv,
                            float* __restrict__ norm, int E) {
  int e = blockIdx.x * blockDim.x + threadIdx.x;
  if (e < E) {
    int s = (int)ei[e];
    int d = (int)ei[(size_t)E + e];
    norm[e] = dinv[s] * dinv[d];
  }
}

// Build a 16x32 bf16 A fragment (contiguous k = 16*h + s per lane half, as B) from 4 float4 K-blocks.
__device__ __forceinline__ v16bf make_a_frag(float4 f0, float4 f1, float4 f2, float4 f3) {
  v16bf a;
  a[0]=(bf16_t)f0.x;  a[1]=(bf16_t)f0.y;  a[2]=(bf16_t)f0.z;  a[3]=(bf16_t)f0.w;
  a[4]=(bf16_t)f1.x;  a[5]=(bf16_t)f1.y;  a[6]=(bf16_t)f1.z;  a[7]=(bf16_t)f1.w;
  a[8]=(bf16_t)f2.x;  a[9]=(bf16_t)f2.y;  a[10]=(bf16_t)f2.z; a[11]=(bf16_t)f2.w;
  a[12]=(bf16_t)f3.x; a[13]=(bf16_t)f3.y; a[14]=(bf16_t)f3.z; a[15]=(bf16_t)f3.w;
  return a;
}

// ---------------------------------------------------------------------------
// GEMM1: h1[N,256] = x[N,512] @ W1 ; bf16 WMMA, f32 accum.
// One wave per block; wave owns 16-row tile x 4 column tiles (grid.y picks which 4).
// All B fragments for a K-step are issued before the WMMA burst so the
// loadcnt waits stage down instead of draining per-WMMA.
// ---------------------------------------------------------------------------
__global__ __launch_bounds__(32)
void gemm1_kernel(const float* __restrict__ x, const bf16_t* __restrict__ w1t,
                  float* __restrict__ h1, int N) {
  const int lane = threadIdx.x;
  const int nodeTile = blockIdx.x;
  const int ctBase = blockIdx.y * 4;
  const int nlane = lane & 15;

  int row = nodeTile * 16 + nlane;
  if (row >= N) row = N - 1;                 // clamp loads (stores guarded)
  const float* xrow = x + (size_t)row * FAN_IN;

  const int aoff = (lane < 16) ? 0 : 16;     // A: K base per lane group
  const int boff = (lane < 16) ? 0 : 16;     // B: K base per lane group

  // Per-column-tile B row pointers (column-major-transposed weights)
  const bf16_t* bp[4];
#pragma unroll
  for (int t = 0; t < 4; t++)
    bp[t] = w1t + (size_t)((ctBase + t) * 16 + nlane) * FAN_IN + boff;

  v8f acc[4] = {};

  for (int k0 = 0; k0 < FAN_IN; k0 += 32) {
    // A loads (4 x b128), issued first
    float4 f0 = *(const float4*)(xrow + k0 + aoff + 0);
    float4 f1 = *(const float4*)(xrow + k0 + aoff + 4);
    float4 f2 = *(const float4*)(xrow + k0 + aoff + 8);
    float4 f3 = *(const float4*)(xrow + k0 + aoff + 12);

    // Prefetch next K-block of the A row while this step computes
    if (k0 + 32 < FAN_IN)
      __builtin_prefetch(xrow + k0 + 32 + aoff, 0, 0);

    // All B fragments (8 x b128) into distinct registers
    v16bf b0 = *(const v16bf*)(bp[0] + k0);
    v16bf b1 = *(const v16bf*)(bp[1] + k0);
    v16bf b2 = *(const v16bf*)(bp[2] + k0);
    v16bf b3 = *(const v16bf*)(bp[3] + k0);

    v16bf a = make_a_frag(f0, f1, f2, f3);

    acc[0] = __builtin_amdgcn_wmma_f32_16x16x32_bf16(false, a, false, b0, (short)0, acc[0], false, false);
    acc[1] = __builtin_amdgcn_wmma_f32_16x16x32_bf16(false, a, false, b1, (short)0, acc[1], false, false);
    acc[2] = __builtin_amdgcn_wmma_f32_16x16x32_bf16(false, a, false, b2, (short)0, acc[2], false, false);
    acc[3] = __builtin_amdgcn_wmma_f32_16x16x32_bf16(false, a, false, b3, (short)0, acc[3], false, false);
  }

  const int mbase = nodeTile * 16 + ((lane < 16) ? 0 : 8);
#pragma unroll
  for (int t = 0; t < 4; t++) {
    int col = (ctBase + t) * 16 + nlane;
#pragma unroll
    for (int v = 0; v < 8; v++) {
      int m = mbase + v;
      if (m < N) h1[(size_t)m * FAN_MID + col] = acc[t][v];
    }
  }
}

// ---------------------------------------------------------------------------
// Edge aggregation, layer 1: one block (256 threads) per edge, thread = feature.
// ---------------------------------------------------------------------------
__global__ __launch_bounds__(256)
void agg1_kernel(const long long* __restrict__ ei, const float* __restrict__ norm,
                 const float* __restrict__ h1, float* __restrict__ agg1, int E) {
  int e = blockIdx.x;
  int f = threadIdx.x;
  int s = (int)ei[e];
  int d = (int)ei[(size_t)E + e];
  float w = norm[e];
  atomicAdd(&agg1[(size_t)d * FAN_MID + f], h1[(size_t)s * FAN_MID + f] * w);
}

// ---------------------------------------------------------------------------
// LN1 + ELU: one wave per node (8 waves/block). Adds self-loop term + bias.
// hact may alias h1 (each (n,f) read before written by the owning thread).
// ---------------------------------------------------------------------------
__global__ __launch_bounds__(256)
void ln1_kernel(const float* agg1, const float* h1, float* hact,
                const float* __restrict__ dinv, const float* __restrict__ b1,
                const float* __restrict__ g1, const float* __restrict__ be1, int N) {
  int wave = threadIdx.x >> 5;
  int lane = threadIdx.x & 31;
  int n = blockIdx.x * 8 + wave;
  if (n >= N) return;

  float di = dinv[n];
  float d2 = di * di;                        // = 1/deg
  const float* ar = agg1 + (size_t)n * FAN_MID;
  const float* hr = h1   + (size_t)n * FAN_MID;

  float v[8], s = 0.0f, ss = 0.0f;
#pragma unroll
  for (int j = 0; j < 8; j++) {
    int f = lane + 32 * j;
    float t = ar[f] + hr[f] * d2 + b1[f];
    v[j] = t; s += t; ss += t * t;
  }
#pragma unroll
  for (int m = 16; m >= 1; m >>= 1) {
    s  += __shfl_xor(s,  m, 32);
    ss += __shfl_xor(ss, m, 32);
  }
  float mu  = s * (1.0f / FAN_MID);
  float var = ss * (1.0f / FAN_MID) - mu * mu;
  float r = rsqrtf(var + LN_EPS);

  float* o = hact + (size_t)n * FAN_MID;
#pragma unroll
  for (int j = 0; j < 8; j++) {
    int f = lane + 32 * j;
    float y = (v[j] - mu) * r * g1[f] + be1[f];
    y = (y > 0.0f) ? y : (expf(y) - 1.0f);   // ELU(alpha=1)
    o[f] = y;
  }
}

// ---------------------------------------------------------------------------
// GEMM2: h2lin[N,40] = hact[N,256] @ W2 ; 3 padded column tiles per wave.
// ---------------------------------------------------------------------------
__global__ __launch_bounds__(32)
void gemm2_kernel(const float* __restrict__ hact, const bf16_t* __restrict__ w2t,
                  float* __restrict__ h2lin, int N) {
  const int lane = threadIdx.x;
  const int nodeTile = blockIdx.x;
  const int nlane = lane & 15;

  int row = nodeTile * 16 + nlane;
  if (row >= N) row = N - 1;
  const float* xrow = hact + (size_t)row * FAN_MID;

  const int aoff = (lane < 16) ? 0 : 16;
  const int boff = (lane < 16) ? 0 : 16;

  const bf16_t* bp[3];
#pragma unroll
  for (int t = 0; t < 3; t++)
    bp[t] = w2t + (size_t)(t * 16 + nlane) * FAN_MID + boff;

  v8f acc[3] = {};

  for (int k0 = 0; k0 < FAN_MID; k0 += 32) {
    float4 f0 = *(const float4*)(xrow + k0 + aoff + 0);
    float4 f1 = *(const float4*)(xrow + k0 + aoff + 4);
    float4 f2 = *(const float4*)(xrow + k0 + aoff + 8);
    float4 f3 = *(const float4*)(xrow + k0 + aoff + 12);

    if (k0 + 32 < FAN_MID)
      __builtin_prefetch(xrow + k0 + 32 + aoff, 0, 0);

    v16bf b0 = *(const v16bf*)(bp[0] + k0);
    v16bf b1 = *(const v16bf*)(bp[1] + k0);
    v16bf b2 = *(const v16bf*)(bp[2] + k0);

    v16bf a = make_a_frag(f0, f1, f2, f3);

    acc[0] = __builtin_amdgcn_wmma_f32_16x16x32_bf16(false, a, false, b0, (short)0, acc[0], false, false);
    acc[1] = __builtin_amdgcn_wmma_f32_16x16x32_bf16(false, a, false, b1, (short)0, acc[1], false, false);
    acc[2] = __builtin_amdgcn_wmma_f32_16x16x32_bf16(false, a, false, b2, (short)0, acc[2], false, false);
  }

  const int mbase = nodeTile * 16 + ((lane < 16) ? 0 : 8);
#pragma unroll
  for (int t = 0; t < 3; t++) {
    int col = t * 16 + nlane;
    if (col >= FAN_OUT) continue;            // drop padding columns
#pragma unroll
    for (int v = 0; v < 8; v++) {
      int m = mbase + v;
      if (m < N) h2lin[(size_t)m * FAN_OUT + col] = acc[t][v];
    }
  }
}

// ---------------------------------------------------------------------------
// Edge aggregation, layer 2 (40 features): flat thread per (edge, feature)
// ---------------------------------------------------------------------------
__global__ __launch_bounds__(256)
void agg2_kernel(const long long* __restrict__ ei, const float* __restrict__ norm,
                 const float* __restrict__ h2lin, float* __restrict__ agg2, int E) {
  size_t idx = (size_t)blockIdx.x * blockDim.x + threadIdx.x;
  if (idx >= (size_t)E * FAN_OUT) return;
  int e = (int)(idx / FAN_OUT);
  int f = (int)(idx % FAN_OUT);
  int s = (int)ei[e];
  int d = (int)ei[(size_t)E + e];
  atomicAdd(&agg2[(size_t)d * FAN_OUT + f],
            h2lin[(size_t)s * FAN_OUT + f] * norm[e]);
}

// ---------------------------------------------------------------------------
// Final: self-loop + bias, LN, ELU, log_softmax. One thread per node.
// ---------------------------------------------------------------------------
__global__ __launch_bounds__(256)
void final_kernel(const float* __restrict__ agg2, const float* __restrict__ h2lin,
                  const float* __restrict__ dinv, const float* __restrict__ b2,
                  const float* __restrict__ g2, const float* __restrict__ be2,
                  float* __restrict__ out, int N) {
  int n = blockIdx.x * blockDim.x + threadIdx.x;
  if (n >= N) return;
  float di = dinv[n];
  float d2 = di * di;

  float v[FAN_OUT];
  float s = 0.0f, ss = 0.0f;
  for (int f = 0; f < FAN_OUT; f++) {
    float t = agg2[(size_t)n * FAN_OUT + f] + h2lin[(size_t)n * FAN_OUT + f] * d2 + b2[f];
    v[f] = t; s += t; ss += t * t;
  }
  float mu  = s * (1.0f / FAN_OUT);
  float var = ss * (1.0f / FAN_OUT) - mu * mu;
  float r = rsqrtf(var + LN_EPS);

  float mx = -3.4e38f;
  for (int f = 0; f < FAN_OUT; f++) {
    float y = (v[f] - mu) * r * g2[f] + be2[f];
    y = (y > 0.0f) ? y : (expf(y) - 1.0f);
    v[f] = y;
    mx = fmaxf(mx, y);
  }
  float se = 0.0f;
  for (int f = 0; f < FAN_OUT; f++) se += expf(v[f] - mx);
  float lse = mx + logf(se);
  for (int f = 0; f < FAN_OUT; f++) out[(size_t)n * FAN_OUT + f] = v[f] - lse;
}

// ---------------------------------------------------------------------------
extern "C" void kernel_launch(void* const* d_in, const int* in_sizes, int n_in,
                              void* d_out, int out_size, void* d_ws, size_t ws_size,
                              hipStream_t stream) {
  const float*     x   = (const float*)d_in[0];
  const long long* ei  = (const long long*)d_in[1];   // int64 [2, E]
  const float*     W1  = (const float*)d_in[2];
  const float*     b1  = (const float*)d_in[3];
  const float*     g1  = (const float*)d_in[4];
  const float*     be1 = (const float*)d_in[5];
  const float*     W2  = (const float*)d_in[6];
  const float*     b2  = (const float*)d_in[7];
  const float*     g2  = (const float*)d_in[8];
  const float*     be2 = (const float*)d_in[9];
  float* out = (float*)d_out;

  const int N = in_sizes[0] / FAN_IN;    // 50000
  const int E = in_sizes[1] / 2;         // 800000

  // Workspace carve (256B aligned slices)
  size_t off = 0;
  auto carve = [&](size_t bytes) -> char* {
    char* p = (char*)d_ws + off;
    off += (bytes + 255) & ~(size_t)255;
    return p;
  };
  float*  deg   = (float*)carve((size_t)N * sizeof(float));
  float*  dinv  = (float*)carve((size_t)N * sizeof(float));
  float*  norm  = (float*)carve((size_t)E * sizeof(float));
  bf16_t* w1t   = (bf16_t*)carve((size_t)FAN_MID * FAN_IN * sizeof(bf16_t));
  bf16_t* w2t   = (bf16_t*)carve((size_t)FAN_OUT_P * FAN_MID * sizeof(bf16_t));
  float*  h1    = (float*)carve((size_t)N * FAN_MID * sizeof(float));  // also hact
  float*  agg1  = (float*)carve((size_t)N * FAN_MID * sizeof(float));
  float*  h2lin = (float*)carve((size_t)N * FAN_OUT * sizeof(float));
  float*  agg2  = (float*)carve((size_t)N * FAN_OUT * sizeof(float));
  (void)ws_size; (void)n_in; (void)out_size;

  const int nodeTiles = (N + 15) / 16;

  prep_w_kernel<<<(FAN_MID * FAN_IN + 255) / 256, 256, 0, stream>>>(W1, W2, w1t, w2t);

  init_kernel<<<(int)(((size_t)N * FAN_MID + 255) / 256), 256, 0, stream>>>(agg1, agg2, deg, N);
  deg_kernel<<<(E + 255) / 256, 256, 0, stream>>>(ei, deg, E);
  dinv_kernel<<<(N + 255) / 256, 256, 0, stream>>>(deg, dinv, N);
  norm_kernel<<<(E + 255) / 256, 256, 0, stream>>>(ei, dinv, norm, E);

  gemm1_kernel<<<dim3(nodeTiles, 4), 32, 0, stream>>>(x, w1t, h1, N);
  agg1_kernel<<<E, FAN_MID, 0, stream>>>(ei, norm, h1, agg1, E);
  ln1_kernel<<<(N + 7) / 8, 256, 0, stream>>>(agg1, h1, /*hact=*/h1, dinv, b1, g1, be1, N);

  gemm2_kernel<<<nodeTiles, 32, 0, stream>>>(h1, w2t, h2lin, N);
  agg2_kernel<<<(int)(((size_t)E * FAN_OUT + 255) / 256), 256, 0, stream>>>(ei, norm, h2lin, agg2, E);
  final_kernel<<<(N + 255) / 256, 256, 0, stream>>>(agg2, h2lin, dinv, b2, g2, be2, out, N);
}
